// LightningIndexer_75694503625282
// MI455X (gfx1250) — compile-verified, matrix operand order checked by reading, not run on hardware
//
#include <hip/hip_runtime.h>

// ---------------------------------------------------------------------------
// LightningIndexer on MI455X (gfx1250, wave32, WMMA + TDM).
//   qk = hb @ [Wq;Wk]^T   [4096 x 1088] bf16   (TDM -> LDS double-buffered GEMM)
//   I[b,t,s] = sum_h w[h] * relu(q[b,t,h,:] . k[b,s,:])   (bf16 WMMA, f32 accum)
// ---------------------------------------------------------------------------

typedef __attribute__((ext_vector_type(16))) __bf16 v16bf;
typedef __attribute__((ext_vector_type(8)))  __bf16 v8bf;
typedef __attribute__((ext_vector_type(4)))  __bf16 v4bf;
typedef __attribute__((ext_vector_type(8)))  float  v8f;
typedef unsigned int u32x4 __attribute__((ext_vector_type(4)));
typedef int          i32x8 __attribute__((ext_vector_type(8)));
typedef int          i32x4 __attribute__((ext_vector_type(4)));

#define DMODEL 2048
#define NHEAD  16
#define DHEAD  64
#define BATCH  2
#define SEQ    2048
#define NQ     (NHEAD * DHEAD)   // 1024
#define NQK    (NQ + DHEAD)      // 1088 = 17 * 64
#define MROWS  (BATCH * SEQ)     // 4096

// LDS tile geometry: rows of 32 bf16 = 64B, TDM-padded by 16B -> 80B stride
// (spreads rows across banks, keeps 16B alignment for ds_load_b128).
#define LDS_ROW   80
#define A_TILE_B  (128 * LDS_ROW)   // 10240 B
#define B_TILE_B  (64 * LDS_ROW)    //  5120 B
#define B_BASE    (2 * A_TILE_B)    // B buffers start after the two A buffers

// ---------------------------------------------------------------------------
// f32 -> bf16 conversion, 4 elements/thread.
// ---------------------------------------------------------------------------
__global__ void __launch_bounds__(256) cvt_bf16_x4(const float* __restrict__ src,
                                                   __bf16* __restrict__ dst) {
    const int i = (blockIdx.x * 256 + threadIdx.x) * 4;
    const float4 f = *(const float4*)(src + i);
    v4bf o;
    o[0] = (__bf16)f.x; o[1] = (__bf16)f.y; o[2] = (__bf16)f.z; o[3] = (__bf16)f.w;
    *(v4bf*)(dst + i) = o;
}

// ---------------------------------------------------------------------------
// Tensor Data Mover: 2-D tile load (tile_dim0 = 32 bf16 cols, tile_rows rows,
// tensor row stride 2048 bf16) into LDS with 4-DWORD pad every 16 DWORDs.
// D# layout per CDNA5 ISA ch.8 (group0 128b, group1 256b, groups2/3 zero).
// 6-arg builtin flavor: (u32x4 g0, i32x8 g1, i32x4 g2, i32x4 g3, i32x8, i32 cpol)
// ---------------------------------------------------------------------------
__device__ __forceinline__ void tdm_load_2d(unsigned long long gaddr,
                                            unsigned lds_addr,
                                            unsigned tensor_rows,
                                            unsigned tile_rows) {
    u32x4 g0;
    g0[0] = 1u;                                                // count=1, user
    g0[1] = lds_addr;                                          // lds_addr[31:0]
    g0[2] = (unsigned)gaddr;                                   // global_addr[31:0]
    g0[3] = ((unsigned)(gaddr >> 32) & 0x01FFFFFFu)            // global_addr[56:32]
            | 0x80000000u;                                     // type = 2 (image)
    i32x8 g1;
    g1[0] = (1 << 16)        // data_size = 1 -> 2 bytes
          | (1 << 20)        // pad_enable
          | (3 << 22)        // pad_interval = 3 -> every 16 DWORDs (64B row)
          | (3 << 25);       // pad_amount  = 3 -> 4 DWORDs (16B)
    g1[1] = (int)(2048u << 16);          // tensor_dim0 = 2048 (lo16 in hi half)
    g1[2] = (int)(tensor_rows << 16);    // tensor_dim0 hi = 0 | tensor_dim1 lo16
    g1[3] = (int)(32u << 16);            // tensor_dim1 hi = 0 | tile_dim0 = 32
    g1[4] = (int)tile_rows;              // tile_dim1 | tile_dim2 = 0
    g1[5] = 2048;                        // tensor_dim0_stride lo32
    g1[6] = 0;                           // stride0 hi | stride1 lo
    g1[7] = 0;                           // stride1 hi
    i32x4 z4 = {0, 0, 0, 0};
    i32x8 z8 = {0, 0, 0, 0, 0, 0, 0, 0};
    __builtin_amdgcn_tensor_load_to_lds(g0, g1, z4, z4, z8, 0);
}

// ---------------------------------------------------------------------------
// WMMA 16-bit A/B fragment from an LDS tile (LDS_ROW-byte row stride):
//   lane 0..15 : M=lane, K = 0..7 / 16..23   (byte offsets 0 and 32)
//   lane 16..31: M=lane-16, K = 8..15 / 24..31 (byte offsets 16 and 48)
// ---------------------------------------------------------------------------
__device__ __forceinline__ v16bf lds_frag(const unsigned char* p) {
    v8bf a0 = *(const v8bf*)(p);
    v8bf a1 = *(const v8bf*)(p + 32);
    return __builtin_shufflevector(a0, a1, 0, 1, 2, 3, 4, 5, 6, 7,
                                           8, 9, 10, 11, 12, 13, 14, 15);
}

// ---------------------------------------------------------------------------
// Global-memory fragment loader (stage 2; qk is L2-resident).
// ---------------------------------------------------------------------------
__device__ __forceinline__ v16bf load_frag(const __bf16* __restrict__ base,
                                           int ld, int row0, int col0, int lane) {
    const int m  = lane & 15;
    const int ko = (lane >> 4) << 3;
    const __bf16* p = base + (long)(row0 + m) * ld + col0 + ko;
    v8bf a0 = *(const v8bf*)(p);
    v8bf a1 = *(const v8bf*)(p + 16);
    return __builtin_shufflevector(a0, a1, 0, 1, 2, 3, 4, 5, 6, 7,
                                           8, 9, 10, 11, 12, 13, 14, 15);
}

// ---------------------------------------------------------------------------
// GEMM1: qk[4096 x 1088] (bf16) = hb[4096 x 2048] @ wb[1088 x 2048]^T
// Block (8 waves) -> 128(M) x 64(N); TDM double-buffers A(128x32), B(64x32)
// in LDS. Wave w computes rows m0+16w, 4 N-subtiles. 544 blocks (32 x 17).
// ---------------------------------------------------------------------------
__global__ void __launch_bounds__(256) gemm_qk(const __bf16* __restrict__ hb,
                                               const __bf16* __restrict__ wb,
                                               __bf16* __restrict__ qk) {
    __shared__ __align__(16) unsigned char smem[2 * A_TILE_B + 2 * B_TILE_B];

    const int lane = threadIdx.x & 31;
    const int wv   = threadIdx.x >> 5;
    const int m0   = (blockIdx.x / 17) * 128;
    const int n0   = (blockIdx.x % 17) * 64;

    const unsigned long long hA = (unsigned long long)(uintptr_t)hb;
    const unsigned long long hB = (unsigned long long)(uintptr_t)wb;
    // Flat shared pointer: low 32 bits are the LDS byte address (LDS aperture).
    const unsigned ldsBase = (unsigned)(uintptr_t)(void*)smem;

    // Per-lane fragment offsets within a tile (LDS_ROW-byte padded rows).
    const int fm  = lane & 15;
    const int fko = (lane >> 4) << 4;                 // 0 or 16 bytes
    const int afrag = (wv * 16 + fm) * LDS_ROW + fko; // this wave's A rows

    // Prologue: DMA k-step 0 into buffer 0.
    if (wv == 0) {
        tdm_load_2d(hA + (unsigned long long)m0 * DMODEL * 2,
                    ldsBase + 0, MROWS, 128);
        tdm_load_2d(hB + (unsigned long long)n0 * DMODEL * 2,
                    ldsBase + B_BASE, NQK, 64);
    }

    v8f acc[4] = {};
    for (int kk = 0; kk < DMODEL; kk += 32) {
        const int buf = (kk >> 5) & 1;
        const bool has_next = (kk + 32) < DMODEL;
        if (wv == 0) {
            if (has_next) {
                // Other buffer was fully consumed before the barrier that
                // ended the previous iteration -> safe to overwrite.
                const int nb = buf ^ 1;
                tdm_load_2d(hA + ((unsigned long long)m0 * DMODEL + kk + 32) * 2,
                            ldsBase + nb * A_TILE_B, MROWS, 128);
                tdm_load_2d(hB + ((unsigned long long)n0 * DMODEL + kk + 32) * 2,
                            ldsBase + B_BASE + nb * B_TILE_B, NQK, 64);
                __builtin_amdgcn_s_wait_tensorcnt(2);  // current pair done
            } else {
                __builtin_amdgcn_s_wait_tensorcnt(0);
            }
        }
        __syncthreads();  // tile `buf` visible to all waves

        const unsigned char* aP = smem + buf * A_TILE_B + afrag;
        const unsigned char* bP = smem + B_BASE + buf * B_TILE_B + fm * LDS_ROW + fko;
        v16bf a = lds_frag(aP);
#pragma unroll
        for (int n = 0; n < 4; ++n) {
            v16bf b = lds_frag(bP + n * 16 * LDS_ROW);
            acc[n] = __builtin_amdgcn_wmma_f32_16x16x32_bf16(
                false, a, false, b, (short)0, acc[n], false, false);
        }
        __syncthreads();  // all reads of `buf` done before next overwrite
    }

    // C/D layout: lane holds column n = lane&15, rows mb..mb+7, mb=(lane>>4)*8
    const int n  = lane & 15;
    const int mb = (lane >> 4) << 3;
    const int row0 = m0 + wv * 16;
#pragma unroll
    for (int t = 0; t < 4; ++t) {
#pragma unroll
        for (int r = 0; r < 8; ++r) {
            qk[(long)(row0 + mb + r) * NQK + n0 + t * 16 + n] = (__bf16)acc[t][r];
        }
    }
}

// ---------------------------------------------------------------------------
// Stage 2: I[b,t,s] = sum_h w[h] * relu( q_h[t,:] . k[s,:] )
// One wave -> 16(T) x 64(S); K fragments hoisted across all 16 heads.
// 2 x 128 x 32 = 8192 waves = 1024 blocks x 8 waves.
// ---------------------------------------------------------------------------
__global__ void __launch_bounds__(256) lightning_scores(const __bf16* __restrict__ qk,
                                                        const float* __restrict__ w,
                                                        float* __restrict__ out) {
    const int lane = threadIdx.x & 31;
    const int gw   = blockIdx.x * 8 + (threadIdx.x >> 5);
    const int sg   = gw % 32;
    const int tt   = (gw / 32) % 128;
    const int b    = gw / (32 * 128);
    const int trow0 = b * SEQ + tt * 16;
    const int srow0 = b * SEQ + sg * 64;

    v16bf kf[4][2];
#pragma unroll
    for (int nIt = 0; nIt < 4; ++nIt) {
        kf[nIt][0] = load_frag(qk, NQK, srow0 + nIt * 16, NQ + 0,  lane);
        kf[nIt][1] = load_frag(qk, NQK, srow0 + nIt * 16, NQ + 32, lane);
    }

    v8f iacc[4] = {};
    for (int hh = 0; hh < NHEAD; ++hh) {
        v16bf a0 = load_frag(qk, NQK, trow0, hh * DHEAD + 0,  lane);
        v16bf a1 = load_frag(qk, NQK, trow0, hh * DHEAD + 32, lane);
        const float wh = w[hh];
#pragma unroll
        for (int nIt = 0; nIt < 4; ++nIt) {
            v8f acc = {};
            acc = __builtin_amdgcn_wmma_f32_16x16x32_bf16(
                false, a0, false, kf[nIt][0], (short)0, acc, false, false);
            acc = __builtin_amdgcn_wmma_f32_16x16x32_bf16(
                false, a1, false, kf[nIt][1], (short)0, acc, false, false);
#pragma unroll
            for (int r = 0; r < 8; ++r) {
                iacc[nIt][r] += wh * fmaxf(acc[r], 0.0f);
            }
        }
    }

    const int n  = lane & 15;
    const int mb = (lane >> 4) << 3;
    const int scol0 = sg * 64;
#pragma unroll
    for (int t = 0; t < 4; ++t) {
#pragma unroll
        for (int r = 0; r < 8; ++r) {
            out[(long)(trow0 + mb + r) * SEQ + scol0 + t * 16 + n] = iacc[t][r];
        }
    }
}

// ---------------------------------------------------------------------------
// Launch
// ---------------------------------------------------------------------------
extern "C" void kernel_launch(void* const* d_in, const int* in_sizes, int n_in,
                              void* d_out, int out_size, void* d_ws, size_t ws_size,
                              hipStream_t stream) {
    const float* h   = (const float*)d_in[0];   // [2,2048,2048]
    const float* w_q = (const float*)d_in[1];   // [1024,2048]
    const float* w_k = (const float*)d_in[2];   // [64,2048]
    const float* w   = (const float*)d_in[3];   // [16]
    float* out = (float*)d_out;                 // [2,2048,2048]

    char* ws = (char*)d_ws;
    __bf16* hb = (__bf16*)ws;                                          // 4096x2048 bf16
    __bf16* wb = (__bf16*)(ws + (size_t)MROWS * DMODEL * 2);           // 1088x2048 bf16
    __bf16* qk = (__bf16*)(ws + (size_t)MROWS * DMODEL * 2
                              + (size_t)NQK * DMODEL * 2);             // 4096x1088 bf16

    cvt_bf16_x4<<<(MROWS * DMODEL) / 1024, 256, 0, stream>>>(h, hb);
    cvt_bf16_x4<<<(NQ * DMODEL) / 1024, 256, 0, stream>>>(w_q, wb);
    cvt_bf16_x4<<<(DHEAD * DMODEL) / 1024, 256, 0, stream>>>(w_k, wb + (size_t)NQ * DMODEL);

    gemm_qk<<<544, 256, 0, stream>>>(hb, wb, qk);
    lightning_scores<<<1024, 256, 0, stream>>>(qk, w, out);
}
